// BottleneckAttention_27891517621088
// MI455X (gfx1250) — matrix operand order repaired and checked by reading, not compiled, not run on hardware
//
#include <hip/hip_runtime.h>
#include <hip/hip_bf16.h>

// ---------------------------------------------------------------------------
// BottleneckAttention for MI455X (gfx1250, wave32, WMMA 16x16x32 f16).
// B=2, N=2048, D=1024, H=16, hd=64.
// Data movement uses the CDNA5-specific paths: TDM (tensor_load_to_lds) for
// the contiguous GEMM A-panels, async global->LDS b128 for attention K tiles.
// ---------------------------------------------------------------------------

#define DMODEL 1024
#define NSEQ   2048
#define NROWS  4096      // B*N
#define NHEADS 16
#define HDIM   64

typedef __attribute__((ext_vector_type(16))) _Float16     v16h;
typedef __attribute__((ext_vector_type(8)))  float        v8f;
typedef __attribute__((ext_vector_type(4)))  unsigned int v4u;
typedef __attribute__((ext_vector_type(8)))  int          v8i;
typedef __attribute__((ext_vector_type(4)))  int          v4i;

union Frag16 { v16h h; uint4 u[2]; };

// ---- WMMA wrapper: D = A(16x32 f16) * B(32x16 f16) + C(16x16 f32) ----------
__device__ __forceinline__ v8f wmma_f16(v16h a, v16h b, v8f c) {
  return __builtin_amdgcn_wmma_f32_16x16x32_f16(
      /*neg_a=*/false, a, /*neg_b=*/false, b,
      /*c_mod=*/(short)0, c, /*reuse_a=*/false, /*reuse_b=*/false);
}

// A fragment (16x32, MxK), row-major source, rows 0..15 from `base`.
__device__ __forceinline__ v16h frag_a(const _Float16* base, int ld, int k0, int lane) {
  const int r  = lane & 15;
  const int kh = lane >> 4;
  const _Float16* p = base + (size_t)r * ld + k0 + kh * 8;
  Frag16 f;
  f.u[0] = *(const uint4*)(p);       // K = kh*8 + 0..7
  f.u[1] = *(const uint4*)(p + 16);  // K = 16 + kh*8 + 0..7
  return f.h;
}

// B fragment (32x16, KxN) where B[k][n] = src[n][k] (row-major src, ld elems).
__device__ __forceinline__ v16h frag_b(const _Float16* src, int ld, int n0, int k0, int lane) {
  const int c  = lane & 15;
  const int kh = lane >> 4;
  const _Float16* p = src + (size_t)(n0 + c) * ld + k0 + kh * 8;
  Frag16 f;
  f.u[0] = *(const uint4*)(p);
  f.u[1] = *(const uint4*)(p + 16);
  return f.h;
}

// Row reductions across the 16 lanes holding one C-matrix row (wave32).
__device__ __forceinline__ float rowmax16(float v) {
#pragma unroll
  for (int d = 1; d < 16; d <<= 1) v = fmaxf(v, __shfl_xor(v, d, 32));
  return v;
}
__device__ __forceinline__ float rowsum16(float v) {
#pragma unroll
  for (int d = 1; d < 16; d <<= 1) v += __shfl_xor(v, d, 32);
  return v;
}

// ---------------------------------------------------------------------------
// TDM: 2D tile copy global -> LDS (D# groups per CDNA5 ISA).
// data_size_code: 0=1B,1=2B,2=4B,3=8B; dims/strides in data_size units.
// Issue from a single wave; wait with s_wait_tensorcnt.
// ---------------------------------------------------------------------------
__device__ __forceinline__ void tdm_load_2d(unsigned lds_addr, unsigned long long gaddr,
                                            int data_size_code,
                                            unsigned tensor_d0, unsigned tensor_d1,
                                            unsigned tile_d0, unsigned tile_d1,
                                            unsigned stride0) {
  v4u g0;
  g0[0] = 1u;                                   // count=1 (valid), user mode
  g0[1] = lds_addr;                             // lds_addr [63:32]
  g0[2] = (unsigned)gaddr;                      // global_addr [95:64]
  g0[3] = (unsigned)((gaddr >> 32) & 0x01ffffffu) | (2u << 30);  // addr[56:32] | type=2
  v8i g1;
  g1[0] = data_size_code << 16;                 // wg_mask=0, data_size [17:16]
  g1[1] = (int)((tensor_d0 & 0xffffu) << 16);   // tensor_dim0 lo16 -> [63:48]
  g1[2] = (int)((tensor_d0 >> 16) | ((tensor_d1 & 0xffffu) << 16)); // d0 hi | d1 lo
  g1[3] = (int)((tensor_d1 >> 16) | (tile_d0 << 16));              // d1 hi | tile_dim0
  g1[4] = (int)tile_d1;                         // tile_dim1 [143:128], tile_dim2=0
  g1[5] = (int)stride0;                         // tensor_dim0_stride lo32
  g1[6] = 0;
  g1[7] = 0;
  v4i gz = {0, 0, 0, 0};
#if defined(__clang_major__) && (__clang_major__ >= 23)
  v8i gz8 = {0, 0, 0, 0, 0, 0, 0, 0};
  __builtin_amdgcn_tensor_load_to_lds(g0, g1, gz, gz, gz8, 0);
#else
  __builtin_amdgcn_tensor_load_to_lds(g0, g1, gz, gz, 0);
#endif
}

// Async per-lane b128 copy global -> LDS (ASYNCcnt-tracked).
__device__ __forceinline__ void async_load_b128(unsigned lds_off, unsigned long long gaddr) {
  asm volatile("global_load_async_to_lds_b128 %0, %1, off"
               :: "v"(lds_off), "v"(gaddr) : "memory");
}

// ---------------------------------------------------------------------------
// Kernel 0: fp32 -> fp16 weight conversion
// ---------------------------------------------------------------------------
__global__ __launch_bounds__(256) void bna_cvt(const float* __restrict__ s,
                                               _Float16* __restrict__ d, int n4) {
  const int i = blockIdx.x * blockDim.x + threadIdx.x;
  if (i < n4) {
    const float4 v = ((const float4*)s)[i];
    _Float16* o = d + (size_t)i * 4;
    o[0] = (_Float16)v.x; o[1] = (_Float16)v.y;
    o[2] = (_Float16)v.z; o[3] = (_Float16)v.w;
  }
}

// ---------------------------------------------------------------------------
// Kernel 1: LayerNorm (one 256-thread block per row), output f16
// ---------------------------------------------------------------------------
__global__ __launch_bounds__(256) void bna_ln(const float* __restrict__ x,
                                              const float* __restrict__ w,
                                              const float* __restrict__ bb,
                                              _Float16* __restrict__ xh) {
  const int row  = blockIdx.x;
  const int tid  = threadIdx.x;
  const int lane = tid & 31, wave = tid >> 5;
  const float4 v = ((const float4*)(x + (size_t)row * DMODEL))[tid];
  float s1 = v.x + v.y + v.z + v.w;
  float s2 = v.x * v.x + v.y * v.y + v.z * v.z + v.w * v.w;
#pragma unroll
  for (int d = 1; d < 32; d <<= 1) {
    s1 += __shfl_xor(s1, d, 32);
    s2 += __shfl_xor(s2, d, 32);
  }
  __shared__ float r1[8], r2[8];
  if (lane == 0) { r1[wave] = s1; r2[wave] = s2; }
  __syncthreads();
  float t1 = 0.f, t2 = 0.f;
#pragma unroll
  for (int i = 0; i < 8; ++i) { t1 += r1[i]; t2 += r2[i]; }
  const float mu  = t1 * (1.0f / DMODEL);
  const float var = t2 * (1.0f / DMODEL) - mu * mu;
  const float rs  = rsqrtf(var + 1e-5f);
  const float4 wv = ((const float4*)w)[tid];
  const float4 bv = ((const float4*)bb)[tid];
  _Float16* o = xh + (size_t)row * DMODEL + tid * 4;
  o[0] = (_Float16)((v.x - mu) * rs * wv.x + bv.x);
  o[1] = (_Float16)((v.y - mu) * rs * wv.y + bv.y);
  o[2] = (_Float16)((v.z - mu) * rs * wv.z + bv.z);
  o[3] = (_Float16)((v.w - mu) * rs * wv.w + bv.w);
}

// ---------------------------------------------------------------------------
// Kernel 2: Y = X(4096x1024) @ W^T + bias, optional fused RoPE, output f16.
// Block = 128 threads (4 waves); block tile 16 rows x 256 cols; per-wave 16x64.
// A-panel (16x1024 f16 = 32 KB, contiguous) staged by one TDM copy.
// ---------------------------------------------------------------------------
__global__ __launch_bounds__(128) void bna_proj(const _Float16* __restrict__ X,
                                                const _Float16* __restrict__ W,
                                                const float* __restrict__ bias,
                                                _Float16* __restrict__ Y,
                                                int applyRope) {
  __shared__ __align__(16) _Float16 sA[16 * DMODEL];
  const int tid = threadIdx.x, lane = tid & 31, wave = tid >> 5;
  const int row0 = blockIdx.x * 16;
  const int col0 = blockIdx.y * 256 + wave * 64;

  if (wave == 0) {
    // 32 KB contiguous panel: one TDM transfer of 4096 x 8B units.
    tdm_load_2d((unsigned)(size_t)&sA[0],
                (unsigned long long)(size_t)(X + (size_t)row0 * DMODEL),
                /*8B*/3, 4096, 1, 4096, 1, 4096);
    __builtin_amdgcn_s_wait_tensorcnt(0);
  }
  __syncthreads();

  v8f a0 = {}, a1 = {}, a2 = {}, a3 = {};
  for (int k0 = 0; k0 < DMODEL; k0 += 32) {
    if (k0 + 32 < DMODEL)
      __builtin_prefetch(W + (size_t)(col0 + (lane & 15)) * DMODEL + k0 + 32, 0, 1);
    const v16h a = frag_a(sA, DMODEL, k0, lane);
    a0 = wmma_f16(a, frag_b(W, DMODEL, col0,      k0, lane), a0);
    a1 = wmma_f16(a, frag_b(W, DMODEL, col0 + 16, k0, lane), a1);
    a2 = wmma_f16(a, frag_b(W, DMODEL, col0 + 32, k0, lane), a2);
    a3 = wmma_f16(a, frag_b(W, DMODEL, col0 + 48, k0, lane), a3);
  }

  const int n = lane & 15, h2 = lane >> 4;
  v8f acc[4] = {a0, a1, a2, a3};
#pragma unroll
  for (int t = 0; t < 4; ++t) {
#pragma unroll
    for (int r = 0; r < 8; ++r) {
      const int row = row0 + r + 8 * h2;
      const int col = col0 + t * 16 + n;
      float vv = acc[t][r] + bias[col];
      if (applyRope) {
        const float prt = __shfl_xor(vv, 1, 32);       // rotation partner (col^1)
        const int   d   = col & (HDIM - 1);            // dim within head
        const float fi  = (float)(d >> 1);
        // inv_freq = 10000^(-2i/64) = exp(-(2i/64)*ln(10000))
        const float ang = (float)(row & (NSEQ - 1)) *
                          __expf(fi * (-2.0f / HDIM) * 9.210340371976184f);
        const float cs = __cosf(ang), sn = __sinf(ang);
        vv = (d & 1) ? (prt * sn + vv * cs) : (vv * cs - prt * sn);
      }
      Y[(size_t)row * DMODEL + col] = (_Float16)vv;
    }
  }
}

// ---------------------------------------------------------------------------
// Kernel 3: flash attention. Block = 256 threads (8 waves) handles one
// (b, h, 128-q-row) tile; each wave owns 16 q rows. Per 32-key chunk:
// K tile async-copied straight into LDS (ASYNCcnt), V tile transposed via
// registers, online softmax, P re-staged per-wave in LDS for the PV WMMA.
// ---------------------------------------------------------------------------
__global__ __launch_bounds__(256) void bna_attn(const _Float16* __restrict__ Q,
                                                const _Float16* __restrict__ K,
                                                const _Float16* __restrict__ V,
                                                _Float16* __restrict__ O) {
  __shared__ __align__(16) _Float16 sK[32 * HDIM];        // keys x dims
  __shared__ __align__(16) _Float16 sVt[HDIM * 32];       // dims x keys
  __shared__ __align__(16) _Float16 sP[8 * 16 * 32];      // per-wave P tile
  const int tid = threadIdx.x, lane = tid & 31, wave = tid >> 5;
  const int b = blockIdx.z, h = blockIdx.y;
  const int q0 = blockIdx.x * 128 + wave * 16;
  const size_t ho = (size_t)h * HDIM;

  const _Float16* Qb = Q + ((size_t)b * NSEQ + q0) * DMODEL + ho;
  const _Float16* Kb = K + (size_t)b * NSEQ * DMODEL + ho;
  const _Float16* Vb = V + (size_t)b * NSEQ * DMODEL + ho;

  const v16h qa0 = frag_a(Qb, DMODEL, 0,  lane);
  const v16h qa1 = frag_a(Qb, DMODEL, 32, lane);

  v8f o0 = {}, o1 = {}, o2 = {}, o3 = {};
  float mrun[8], lrun[8];
#pragma unroll
  for (int r = 0; r < 8; ++r) { mrun[r] = -3.0e38f; lrun[r] = 0.0f; }

  const int key = tid >> 3;            // 0..31 (staging)
  const int dg  = (tid & 7) * 8;       // dim group (staging)
  const unsigned ldsK = (unsigned)(size_t)&sK[0] + (unsigned)(key * (HDIM * 2) + dg * 2);
  _Float16* myP = sP + wave * (16 * 32);
  const int n = lane & 15, h2 = lane >> 4;

  for (int j = 0; j < NSEQ; j += 32) {
    __syncthreads();
    // K tile: per-lane async b128 copies, no VGPR staging (ASYNCcnt path)
    async_load_b128(ldsK, (unsigned long long)(size_t)(Kb + (size_t)(j + key) * DMODEL + dg));
    // V tile: transpose through registers (dims x keys)
    uint4 vv4 = *(const uint4*)(Vb + (size_t)(j + key) * DMODEL + dg);
    const _Float16* ph = (const _Float16*)&vv4;
#pragma unroll
    for (int e = 0; e < 8; ++e) sVt[(dg + e) * 32 + key] = ph[e];
    asm volatile("s_wait_asynccnt 0x0" ::: "memory");
    __syncthreads();

    // S = Q @ K^T for 16 q-rows x 32 keys (two 16x16 tiles, k-dim = 64)
    v8f s0 = {}, s1 = {};
    s0 = wmma_f16(qa0, frag_b(sK, HDIM, 0,  0,  lane), s0);
    s0 = wmma_f16(qa1, frag_b(sK, HDIM, 0,  32, lane), s0);
    s1 = wmma_f16(qa0, frag_b(sK, HDIM, 16, 0,  lane), s1);
    s1 = wmma_f16(qa1, frag_b(sK, HDIM, 16, 32, lane), s1);

    // online softmax per row; stage P (f16, row-major 16x32) for PV wmma
#pragma unroll
    for (int r = 0; r < 8; ++r) {
      const float v0 = s0[r] * 0.125f;            // 1/sqrt(64)
      const float v1 = s1[r] * 0.125f;
      const float mx   = rowmax16(fmaxf(v0, v1));
      const float mnew = fmaxf(mrun[r], mx);
      const float al   = __expf(mrun[r] - mnew);
      mrun[r] = mnew;
      const float p0 = __expf(v0 - mnew);
      const float p1 = __expf(v1 - mnew);
      lrun[r] = lrun[r] * al + rowsum16(p0 + p1);
      o0[r] *= al; o1[r] *= al; o2[r] *= al; o3[r] *= al;
      const int m = r + 8 * h2;
      myP[m * 32 + n]      = (_Float16)p0;
      myP[m * 32 + 16 + n] = (_Float16)p1;
    }
    asm volatile("s_wait_dscnt 0x0" ::: "memory");  // wave-private P store->load

    // O += P(16x32) @ V(32x64)
    const v16h pa = frag_a(myP, 32, 0, lane);
    o0 = wmma_f16(pa, frag_b(sVt, 32, 0,  0, lane), o0);
    o1 = wmma_f16(pa, frag_b(sVt, 32, 16, 0, lane), o1);
    o2 = wmma_f16(pa, frag_b(sVt, 32, 32, 0, lane), o2);
    o3 = wmma_f16(pa, frag_b(sVt, 32, 48, 0, lane), o3);
  }

#pragma unroll
  for (int r = 0; r < 8; ++r) {
    const float inv = 1.0f / lrun[r];
    const int m = r + 8 * h2;
    _Float16* orow = O + ((size_t)b * NSEQ + q0 + m) * DMODEL + ho;
    orow[n]      = (_Float16)(o0[r] * inv);
    orow[16 + n] = (_Float16)(o1[r] * inv);
    orow[32 + n] = (_Float16)(o2[r] * inv);
    orow[48 + n] = (_Float16)(o3[r] * inv);
  }
}

// ---------------------------------------------------------------------------
// Kernel 4: out = x + A @ Wo^T + bo  (f32 output, residual fused)
// ---------------------------------------------------------------------------
__global__ __launch_bounds__(128) void bna_outproj(const _Float16* __restrict__ A,
                                                   const _Float16* __restrict__ W,
                                                   const float* __restrict__ bias,
                                                   const float* __restrict__ xres,
                                                   float* __restrict__ Y) {
  __shared__ __align__(16) _Float16 sA[16 * DMODEL];
  const int tid = threadIdx.x, lane = tid & 31, wave = tid >> 5;
  const int row0 = blockIdx.x * 16;
  const int col0 = blockIdx.y * 256 + wave * 64;

  if (wave == 0) {
    tdm_load_2d((unsigned)(size_t)&sA[0],
                (unsigned long long)(size_t)(A + (size_t)row0 * DMODEL),
                /*8B*/3, 4096, 1, 4096, 1, 4096);
    __builtin_amdgcn_s_wait_tensorcnt(0);
  }
  __syncthreads();

  v8f a0 = {}, a1 = {}, a2 = {}, a3 = {};
  for (int k0 = 0; k0 < DMODEL; k0 += 32) {
    if (k0 + 32 < DMODEL)
      __builtin_prefetch(W + (size_t)(col0 + (lane & 15)) * DMODEL + k0 + 32, 0, 1);
    const v16h a = frag_a(sA, DMODEL, k0, lane);
    a0 = wmma_f16(a, frag_b(W, DMODEL, col0,      k0, lane), a0);
    a1 = wmma_f16(a, frag_b(W, DMODEL, col0 + 16, k0, lane), a1);
    a2 = wmma_f16(a, frag_b(W, DMODEL, col0 + 32, k0, lane), a2);
    a3 = wmma_f16(a, frag_b(W, DMODEL, col0 + 48, k0, lane), a3);
  }

  const int n = lane & 15, h2 = lane >> 4;
  v8f acc[4] = {a0, a1, a2, a3};
#pragma unroll
  for (int t = 0; t < 4; ++t) {
#pragma unroll
    for (int r = 0; r < 8; ++r) {
      const int row = row0 + r + 8 * h2;
      const int col = col0 + t * 16 + n;
      const size_t idx = (size_t)row * DMODEL + col;
      Y[idx] = acc[t][r] + bias[col] + xres[idx];
    }
  }
}

// ---------------------------------------------------------------------------
// Host launcher
// ---------------------------------------------------------------------------
extern "C" void kernel_launch(void* const* d_in, const int* in_sizes, int n_in,
                              void* d_out, int out_size, void* d_ws, size_t ws_size,
                              hipStream_t stream) {
  (void)in_sizes; (void)n_in; (void)out_size; (void)ws_size;
  const float* x    = (const float*)d_in[0];
  const float* ln_w = (const float*)d_in[1];
  const float* ln_b = (const float*)d_in[2];
  const float* Wq   = (const float*)d_in[3];
  const float* bq   = (const float*)d_in[4];
  const float* Wk   = (const float*)d_in[5];
  const float* bk   = (const float*)d_in[6];
  const float* Wv   = (const float*)d_in[7];
  const float* bv   = (const float*)d_in[8];
  const float* Wo   = (const float*)d_in[9];
  const float* bo   = (const float*)d_in[10];

  const size_t RD = (size_t)NROWS * DMODEL;   // 4096*1024
  const size_t WD = (size_t)DMODEL * DMODEL;  // 1024*1024
  _Float16* xh  = (_Float16*)d_ws;
  _Float16* Wqh = xh  + RD;
  _Float16* Wkh = Wqh + WD;
  _Float16* Wvh = Wkh + WD;
  _Float16* Woh = Wvh + WD;
  _Float16* Qh  = Woh + WD;
  _Float16* Kh  = Qh  + RD;
  _Float16* Vh  = Kh  + RD;
  _Float16* Ah  = Vh  + RD;

  const int n4 = (int)(WD / 4);
  bna_cvt<<<n4 / 256, 256, 0, stream>>>(Wq, Wqh, n4);
  bna_cvt<<<n4 / 256, 256, 0, stream>>>(Wk, Wkh, n4);
  bna_cvt<<<n4 / 256, 256, 0, stream>>>(Wv, Wvh, n4);
  bna_cvt<<<n4 / 256, 256, 0, stream>>>(Wo, Woh, n4);

  bna_ln<<<NROWS, 256, 0, stream>>>(x, ln_w, ln_b, xh);

  dim3 gg(NROWS / 16, DMODEL / 256);
  bna_proj<<<gg, 128, 0, stream>>>(xh, Wqh, bq, Qh, 1);
  bna_proj<<<gg, 128, 0, stream>>>(xh, Wkh, bk, Kh, 1);
  bna_proj<<<gg, 128, 0, stream>>>(xh, Wvh, bv, Vh, 0);

  dim3 ga(NSEQ / 128, NHEADS, 2);
  bna_attn<<<ga, 256, 0, stream>>>(Qh, Kh, Vh, Ah);

  bna_outproj<<<gg, 128, 0, stream>>>(Ah, Woh, bo, x, (float*)d_out);
}
